// ResNetMamba_6597069767183
// MI455X (gfx1250) — matrix operand order repaired and checked by reading, not compiled, not run on hardware
//
#include <hip/hip_runtime.h>
#include <hip/hip_bf16.h>

typedef _Float16 h16;
typedef __attribute__((ext_vector_type(8)))  _Float16 v8h;
typedef __attribute__((ext_vector_type(16))) _Float16 v16h;
typedef __attribute__((ext_vector_type(8)))  float    v8f;

#define EPSV 1e-5f

// ---------------------------------------------------------------------------
// Generic 1D convolution (NCH, OIH), stride/pad, full Cin reduction.
// ---------------------------------------------------------------------------
__global__ void conv1d_k(const float* __restrict__ in, const float* __restrict__ w,
                         float* __restrict__ out,
                         int B, int Cin, int Cout, int Lin, int Lout,
                         int K, int stride, int padL)
{
  int idx = blockIdx.x * blockDim.x + threadIdx.x;
  int total = B * Cout * Lout;
  if (idx >= total) return;
  int l = idx % Lout;
  int t = idx / Lout;
  int co = t % Cout;
  int b  = t / Cout;
  int l0 = l * stride - padL;
  float acc = 0.f;
  for (int ci = 0; ci < Cin; ++ci) {
    const float* ip = in + ((size_t)b * Cin + ci) * Lin;
    const float* wp = w  + ((size_t)co * Cin + ci) * K;
    for (int k = 0; k < K; ++k) {
      int li = l0 + k;
      if (li >= 0 && li < Lin) acc += wp[k] * ip[li];
    }
  }
  out[idx] = acc;
}

// ---------------------------------------------------------------------------
// BatchNorm batch statistics: one block per channel. mv[c]=mean, mv[C+c]=var.
// ---------------------------------------------------------------------------
__global__ void bn_stats_k(const float* __restrict__ x, float* __restrict__ mv,
                           int B, int C, int L)
{
  int c = blockIdx.x;
  int n = B * L;
  float s = 0.f, s2 = 0.f;
  for (int i = threadIdx.x; i < n; i += blockDim.x) {
    int b = i / L, l = i - b * L;
    float v = x[((size_t)b * C + c) * L + l];
    s += v; s2 += v * v;
  }
  __shared__ float sh[256];
  __shared__ float sh2[256];
  int t = threadIdx.x;
  sh[t] = s; sh2[t] = s2;
  __syncthreads();
  for (int st = 128; st > 0; st >>= 1) {
    if (t < st) { sh[t] += sh[t + st]; sh2[t] += sh2[t + st]; }
    __syncthreads();
  }
  if (t == 0) {
    float m = sh[0] / n;
    mv[c] = m;
    mv[C + c] = sh2[0] / n - m * m;
  }
}

// mode: 0=bn, 1=bn+relu, 2=relu(bn+res), 3=gelu(bn)
__global__ void bn_apply_k(const float* __restrict__ x, const float* __restrict__ mv,
                           const float* __restrict__ g, const float* __restrict__ b,
                           const float* __restrict__ res, float* __restrict__ out,
                           int C, int L, int total, int mode)
{
  int idx = blockIdx.x * blockDim.x + threadIdx.x;
  if (idx >= total) return;
  int c = (idx / L) % C;
  float y = (x[idx] - mv[c]) * rsqrtf(mv[C + c] + EPSV) * g[c] + b[c];
  if (mode == 1)      y = fmaxf(y, 0.f);
  else if (mode == 2) y = fmaxf(y + res[idx], 0.f);
  else if (mode == 3) y = 0.5f * y * (1.f + erff(y * 0.70710678118f));
  out[idx] = y;
}

// maxpool window 3 stride 2 pad 1
__global__ void maxpool_k(const float* __restrict__ in, float* __restrict__ out,
                          int B, int C, int Lin, int Lout)
{
  int idx = blockIdx.x * blockDim.x + threadIdx.x;
  int total = B * C * Lout;
  if (idx >= total) return;
  int l = idx % Lout;
  int bc = idx / Lout;
  const float* ip = in + (size_t)bc * Lin;
  float m = -INFINITY;
#pragma unroll
  for (int j = 0; j < 3; ++j) {
    int li = 2 * l - 1 + j;
    if (li >= 0 && li < Lin) m = fmaxf(m, ip[li]);
  }
  out[idx] = m;
}

// (B,C,L) -> (B,L,C)
__global__ void transpose_k(const float* __restrict__ in, float* __restrict__ out,
                            int B, int C, int L)
{
  int idx = blockIdx.x * blockDim.x + threadIdx.x;
  int total = B * L * C;
  if (idx >= total) return;
  int c = idx % C;
  int l = (idx / C) % L;
  int b = idx / (C * L);
  out[idx] = in[((size_t)b * C + c) * L + l];
}

// LayerNorm over last dim D (blockDim == D == 128); optional f32 / f16 outputs.
__global__ void layernorm_k(const float* __restrict__ x, const float* __restrict__ g,
                            const float* __restrict__ b, float* __restrict__ of32,
                            h16* __restrict__ of16, int D)
{
  int row = blockIdx.x;
  int t = threadIdx.x;
  float v = x[(size_t)row * D + t];
  __shared__ float sm[128];
  __shared__ float sv[128];
  sm[t] = v; sv[t] = v * v;
  __syncthreads();
  for (int s = 64; s > 0; s >>= 1) {
    if (t < s) { sm[t] += sm[t + s]; sv[t] += sv[t + s]; }
    __syncthreads();
  }
  float m = sm[0] / D;
  float var = sv[0] / D - m * m;
  float y = (v - m) * rsqrtf(var + EPSV) * g[t] + b[t];
  if (of32) of32[(size_t)row * D + t] = y;
  if (of16) of16[(size_t)row * D + t] = (h16)y;
}

// ---------------------------------------------------------------------------
// WMMA f16 GEMM:  C[M,N] (f32) = A[M,K] (f16 row-major) x W[N,K]^T (f16 row-major)
// One wave per 16x16 tile of C. K % 32 == 0, M,N % 16 == 0. ldc == N.
// A-fragment per documented 16-bit A 16x32 layout; lanes g=lane>>4:
//   elems 0..7  -> K = k0 + 8*g + i        (v8h load)
//   elems 8..15 -> K = k0 + 16 + 8*g + i-8 (v8h load)
// B-fragment: lane column c = lane&15, same K mapping as A (two v8h loads
// from W[c][...]).
// ---------------------------------------------------------------------------
__global__ void gemm_f16_wmma_k(const h16* __restrict__ A, const h16* __restrict__ W,
                                float* __restrict__ C, int M, int N, int K)
{
  int lane = threadIdx.x & 31;
  int wave = threadIdx.x >> 5;
  int tiles_n = N >> 4;
  int tiles_m = M >> 4;
  int tile = blockIdx.x * (blockDim.x >> 5) + wave;
  if (tile >= tiles_m * tiles_n) return;   // wave-uniform -> EXEC stays all-ones
  int tm = tile / tiles_n;
  int tn = tile - tm * tiles_n;
  int r = lane & 15;
  int g = lane >> 4;
  const h16* arow = A + (size_t)(tm * 16 + r) * K;
  const h16* wrow = W + (size_t)(tn * 16 + r) * K;
  v8f acc = {};
  for (int k0 = 0; k0 < K; k0 += 32) {
    v8h alo = *(const v8h*)(arow + k0 + g * 8);
    v8h ahi = *(const v8h*)(arow + k0 + 16 + g * 8);
    v16h af = __builtin_shufflevector(alo, ahi,
                0, 1, 2, 3, 4, 5, 6, 7, 8, 9, 10, 11, 12, 13, 14, 15);
    v8h blo = *(const v8h*)(wrow + k0 + g * 8);
    v8h bhi = *(const v8h*)(wrow + k0 + 16 + g * 8);
    v16h bf = __builtin_shufflevector(blo, bhi,
                0, 1, 2, 3, 4, 5, 6, 7, 8, 9, 10, 11, 12, 13, 14, 15);
    acc = __builtin_amdgcn_wmma_f32_16x16x32_f16(false, af, false, bf,
                                                 (short)0, acc, false, false);
  }
  // C/D layout: col = lane&15, rows m = tm*16 + 8*g + v
  float* cp = C + (size_t)(tm * 16 + g * 8) * N + tn * 16 + r;
#pragma unroll
  for (int v = 0; v < 8; ++v) cp[(size_t)v * N] = acc[v];
}

// f32 -> f16 copy
__global__ void f2h_k(const float* __restrict__ in, h16* __restrict__ out, int n)
{
  int i = blockIdx.x * blockDim.x + threadIdx.x;
  if (i < n) out[i] = (h16)in[i];
}

// f32 (rows x cols) -> f16 (rows_pad x cols), zero-padded rows
__global__ void f2h_pad_k(const float* __restrict__ in, h16* __restrict__ out,
                          int rows, int cols, int rows_pad)
{
  int i = blockIdx.x * blockDim.x + threadIdx.x;
  if (i >= rows_pad * cols) return;
  int r = i / cols, c = i - r * cols;
  out[i] = (r < rows) ? (h16)in[(size_t)r * cols + c] : (h16)0.f;
}

// A = -exp(A_log)
__global__ void aprep_k(const float* __restrict__ al, float* __restrict__ A, int n)
{
  int i = blockIdx.x * blockDim.x + threadIdx.x;
  if (i < n) A[i] = -expf(al[i]);
}

// Depthwise causal conv (k=4, pad left 3) on xi part of xz (ld 512) + SiLU.
// Writes f32 (for scan) and f16 (x_proj GEMM operand). Layout (B*L, 256).
__global__ void dwconv_silu_k(const float* __restrict__ xz, const float* __restrict__ cw,
                              const float* __restrict__ cb, float* __restrict__ xi,
                              h16* __restrict__ xi16, int L)
{
  int idx = blockIdx.x * blockDim.x + threadIdx.x;
  int total = 8 * L * 256;
  if (idx >= total) return;
  int d = idx & 255;
  int l = (idx >> 8) % L;
  int b = idx / (256 * L);
  float acc = cb[d];
#pragma unroll
  for (int j = 0; j < 4; ++j) {
    int li = l + j - 3;
    if (li >= 0) acc += cw[d * 4 + j] * xz[((size_t)(b * L + li)) * 512 + d];
  }
  float s = acc / (1.f + expf(-acc));
  xi[idx] = s;
  xi16[idx] = (h16)s;
}

// dt = softplus(dbl[:, :8] @ dt_w^T + dt_b), dbl ld 48
__global__ void dt_k(const float* __restrict__ dbl, const float* __restrict__ dw,
                     const float* __restrict__ db, float* __restrict__ dt, int rows)
{
  int idx = blockIdx.x * blockDim.x + threadIdx.x;
  if (idx >= rows * 256) return;
  int d = idx & 255;
  int row = idx >> 8;
  const float* r = dbl + (size_t)row * 48;
  float a = db[d];
#pragma unroll
  for (int j = 0; j < 8; ++j) a += r[j] * dw[d * 8 + j];
  dt[idx] = (a > 20.f) ? a : log1pf(expf(a));
}

// Selective scan fused with skip (x*D) and silu(z) gate. One thread per (b,d),
// 16 SSM states in registers, L sequential steps. Writes f16 out_proj operand.
// dbl ld 48: B at cols 8..23, C at cols 24..39. z from xz (ld 512, offset 256).
__global__ void scan_gate_k(const float* __restrict__ xi, const float* __restrict__ dt,
                            const float* __restrict__ dbl, const float* __restrict__ A,
                            const float* __restrict__ Dp, const float* __restrict__ xz,
                            h16* __restrict__ yg, int L)
{
  int d = threadIdx.x;   // 0..255
  int b = blockIdx.x;    // 0..7
  float h[16];
  float Ad[16];
#pragma unroll
  for (int s = 0; s < 16; ++s) { h[s] = 0.f; Ad[s] = A[d * 16 + s]; }
  float Dv = Dp[d];
  for (int l = 0; l < L; ++l) {
    size_t row = (size_t)b * L + l;
    float x   = xi[row * 256 + d];
    float dtv = dt[row * 256 + d];
    const float* bc = dbl + row * 48;
    float acc = 0.f;
#pragma unroll
    for (int s = 0; s < 16; ++s) {
      float hn = h[s] * expf(dtv * Ad[s]) + dtv * bc[8 + s] * x;
      h[s] = hn;
      acc += hn * bc[24 + s];
    }
    float z = xz[row * 512 + 256 + d];
    float v = (acc + x * Dv) * (z / (1.f + expf(-z)));
    yg[row * 256 + d] = (h16)v;
  }
}

__global__ void addf_k(float* __restrict__ t, const float* __restrict__ a, int n)
{
  int i = blockIdx.x * blockDim.x + threadIdx.x;
  if (i < n) t[i] += a[i];
}

__global__ void add_flip_k(float* __restrict__ t, const float* __restrict__ a,
                           int B, int L, int D)
{
  int idx = blockIdx.x * blockDim.x + threadIdx.x;
  int total = B * L * D;
  if (idx >= total) return;
  int d = idx % D;
  int l = (idx / D) % L;
  int b = idx / (D * L);
  t[idx] += a[(((size_t)b * L) + (L - 1 - l)) * D + d];
}

__global__ void flip_h_k(const h16* __restrict__ in, h16* __restrict__ out,
                         int B, int L, int D)
{
  int idx = blockIdx.x * blockDim.x + threadIdx.x;
  int total = B * L * D;
  if (idx >= total) return;
  int d = idx % D;
  int l = (idx / D) % L;
  int b = idx / (D * L);
  out[idx] = in[(((size_t)b * L) + (L - 1 - l)) * D + d];
}

__global__ void meanpool_k(const float* __restrict__ x, float* __restrict__ out,
                           int B, int L, int D)
{
  int idx = blockIdx.x * blockDim.x + threadIdx.x;
  if (idx >= B * D) return;
  int d = idx % D, b = idx / D;
  float s = 0.f;
  for (int l = 0; l < L; ++l) s += x[((size_t)b * L + l) * D + d];
  out[idx] = s / L;
}

// classifier: gelu(pooled @ w1^T + b1) @ w2^T + b2; grid=B, block=64
__global__ void cls_k(const float* __restrict__ pooled,
                      const float* __restrict__ w1, const float* __restrict__ b1,
                      const float* __restrict__ w2, const float* __restrict__ b2,
                      float* __restrict__ out)
{
  int b = blockIdx.x;
  int j = threadIdx.x;   // 0..63
  __shared__ float h1[64];
  float a = b1[j];
  for (int k = 0; k < 128; ++k) a += pooled[b * 128 + k] * w1[j * 128 + k];
  h1[j] = 0.5f * a * (1.f + erff(a * 0.70710678118f));
  __syncthreads();
  if (j < 2) {
    float o = b2[j];
    for (int k = 0; k < 64; ++k) o += h1[k] * w2[j * 64 + k];
    out[b * 2 + j] = o;
  }
}

// ---------------------------------------------------------------------------
// Host orchestration
// ---------------------------------------------------------------------------
extern "C" void kernel_launch(void* const* d_in, const int* in_sizes, int n_in,
                              void* d_out, int out_size, void* d_ws, size_t ws_size,
                              hipStream_t stream)
{
  (void)in_sizes; (void)n_in; (void)out_size; (void)ws_size;
  auto P = [&](int i) -> const float* { return (const float*)d_in[i]; };
  auto cdiv = [](long a, long b) { return (int)((a + b - 1) / b); };

  const int B = 8;
  const int rows = 8 * 2048;   // 16384 tokens

  char* base = (char*)d_ws;
  size_t off = 0;
  auto allocb = [&](size_t bytes) -> void* {
    void* p = base + off;
    off += (bytes + 255) & ~(size_t)255;
    return p;
  };

  // persistent
  float* mv     = (float*)allocb(512 * sizeof(float));      // bn mean|var
  float* Abuf   = (float*)allocb(256 * 16 * sizeof(float));
  float* pooled = (float*)allocb(B * 128 * sizeof(float));
  float* tbuf   = (float*)allocb((size_t)rows * 128 * sizeof(float));
  size_t scratch0 = off;

  // ---- ResNet scratch ----
  float* s0 = (float*)allocb((size_t)B * 64 * 8192 * sizeof(float));
  float* p0 = (float*)allocb((size_t)B * 128 * 2048 * sizeof(float)); // == 8*64*4096
  float* t1 = (float*)allocb((size_t)B * 128 * 2048 * sizeof(float));
  float* t2 = (float*)allocb((size_t)B * 128 * 2048 * sizeof(float));

  auto conv = [&](const float* in, const float* w, float* out,
                  int Cin, int Cout, int Lin, int Lout, int K, int stride, int pad) {
    long total = (long)B * Cout * Lout;
    conv1d_k<<<cdiv(total, 256), 256, 0, stream>>>(in, w, out, B, Cin, Cout,
                                                   Lin, Lout, K, stride, pad);
  };
  auto stats = [&](const float* x, int C, int L) {
    bn_stats_k<<<C, 256, 0, stream>>>(x, mv, B, C, L);
  };
  auto bnap = [&](const float* x, const float* g, const float* b, const float* res,
                  float* out, int C, int L, int mode) {
    long total = (long)B * C * L;
    bn_apply_k<<<cdiv(total, 256), 256, 0, stream>>>(x, mv, g, b, res, out,
                                                     C, L, (int)total, mode);
  };

  // stem: conv7 s2 p3 -> bn -> relu -> maxpool
  conv(P(0), P(1), s0, 1, 64, 16384, 8192, 7, 2, 3);
  stats(s0, 64, 8192);
  bnap(s0, P(2), P(3), nullptr, s0, 64, 8192, 1);
  maxpool_k<<<cdiv((long)B * 64 * 4096, 256), 256, 0, stream>>>(s0, p0, B, 64, 8192, 4096);

  auto block = [&](float* io, int pb, bool sc, int Cin, int Cout, int Lin, int stride) {
    int Lout = Lin / stride;
    conv(io, P(pb + 0), t1, Cin, Cout, Lin, Lout, 3, stride, 1);
    stats(t1, Cout, Lout);
    bnap(t1, P(pb + 1), P(pb + 2), nullptr, t1, Cout, Lout, 1);
    conv(t1, P(pb + 3), t2, Cout, Cout, Lout, Lout, 3, 1, 1);
    stats(t2, Cout, Lout);
    if (sc) {
      bnap(t2, P(pb + 4), P(pb + 5), nullptr, t2, Cout, Lout, 0);
      conv(io, P(pb + 6), t1, Cin, Cout, Lin, Lout, 1, stride, 0);
      stats(t1, Cout, Lout);
      bnap(t1, P(pb + 7), P(pb + 8), t2, io, Cout, Lout, 2);
    } else {
      bnap(t2, P(pb + 4), P(pb + 5), io, io, Cout, Lout, 2);
    }
  };
  block(p0,  4, false,  64,  64, 4096, 1);   // l1b1
  block(p0, 10, false,  64,  64, 4096, 1);   // l1b2
  block(p0, 16, true,   64, 128, 4096, 2);   // l2b1 (has ws/gs/bs)
  block(p0, 25, false, 128, 128, 2048, 1);   // l2b2

  // proj 1x1 -> bn -> gelu -> transpose to (B,L,D)
  conv(p0, P(31), t1, 128, 128, 2048, 2048, 1, 1, 0);
  stats(t1, 128, 2048);
  bnap(t1, P(32), P(33), nullptr, t1, 128, 2048, 3);
  transpose_k<<<cdiv((long)rows * 128, 256), 256, 0, stream>>>(t1, tbuf, B, 128, 2048);

  // ---- Mamba scratch (re-uses ResNet scratch region; stream is in-order) ----
  off = scratch0;
  float* xz   = (float*)allocb((size_t)rows * 512 * sizeof(float));
  float* xi   = (float*)allocb((size_t)rows * 256 * sizeof(float));
  h16*   xi16 = (h16*)  allocb((size_t)rows * 256 * sizeof(h16));
  float* dblb = (float*)allocb((size_t)rows * 48  * sizeof(float));
  float* dtb  = (float*)allocb((size_t)rows * 256 * sizeof(float));
  h16*   yg16 = (h16*)  allocb((size_t)rows * 256 * sizeof(h16));
  float* mo   = (float*)allocb((size_t)rows * 128 * sizeof(float));
  h16*   tn16 = (h16*)  allocb((size_t)rows * 128 * sizeof(h16));
  h16*   tr16 = (h16*)  allocb((size_t)rows * 128 * sizeof(h16));
  h16*   w16a = (h16*)  allocb((size_t)512 * 128 * sizeof(h16));  // in_proj
  h16*   w16b = (h16*)  allocb((size_t)48  * 256 * sizeof(h16));  // x_proj (padded)
  h16*   w16c = (h16*)  allocb((size_t)128 * 256 * sizeof(h16));  // out_proj

  auto gemm = [&](const h16* A, const h16* W, float* Cc, int M, int N, int K) {
    long tiles = (long)(M / 16) * (N / 16);
    gemm_f16_wmma_k<<<cdiv(tiles, 8), 256, 0, stream>>>(A, W, Cc, M, N, K);
  };
  auto f2h = [&](const float* in, h16* o, long n) {
    f2h_k<<<cdiv(n, 256), 256, 0, stream>>>(in, o, (int)n);
  };

  auto mamba_run = [&](const h16* tnA, int mb) {
    f2h(P(mb + 0), w16a, 512 * 128);
    gemm(tnA, w16a, xz, rows, 512, 128);                             // in_proj (WMMA)
    dwconv_silu_k<<<cdiv((long)rows * 256, 256), 256, 0, stream>>>(
        xz, P(mb + 1), P(mb + 2), xi, xi16, 2048);
    f2h_pad_k<<<cdiv(48 * 256, 256), 256, 0, stream>>>(P(mb + 3), w16b, 40, 256, 48);
    gemm(xi16, w16b, dblb, rows, 48, 256);                           // x_proj (WMMA)
    dt_k<<<cdiv((long)rows * 256, 256), 256, 0, stream>>>(dblb, P(mb + 4), P(mb + 5),
                                                          dtb, rows);
    aprep_k<<<cdiv(4096, 256), 256, 0, stream>>>(P(mb + 6), Abuf, 4096);
    scan_gate_k<<<8, 256, 0, stream>>>(xi, dtb, dblb, Abuf, P(mb + 7), xz, yg16, 2048);
    f2h(P(mb + 8), w16c, 128 * 256);
    gemm(yg16, w16c, mo, rows, 128, 256);                            // out_proj (WMMA)
  };

  for (int blk = 0; blk < 2; ++blk) {
    int bb = 34 + blk * 20;   // ln_g, ln_b, fwd(9), bwd(9)
    layernorm_k<<<rows, 128, 0, stream>>>(tbuf, P(bb), P(bb + 1), nullptr, tn16, 128);
    flip_h_k<<<cdiv((long)rows * 128, 256), 256, 0, stream>>>(tn16, tr16, B, 2048, 128);
    mamba_run(tn16, bb + 2);                                          // fwd
    addf_k<<<cdiv((long)rows * 128, 256), 256, 0, stream>>>(tbuf, mo, rows * 128);
    mamba_run(tr16, bb + 11);                                         // bwd (flipped)
    add_flip_k<<<cdiv((long)rows * 128, 256), 256, 0, stream>>>(tbuf, mo, B, 2048, 128);
  }

  // final LN (f32 into mo) -> mean pool -> classifier
  layernorm_k<<<rows, 128, 0, stream>>>(tbuf, P(74), P(75), mo, nullptr, 128);
  meanpool_k<<<cdiv((long)B * 128, 256), 256, 0, stream>>>(mo, pooled, B, 2048, 128);
  cls_k<<<B, 64, 0, stream>>>(pooled, P(76), P(77), P(78), P(79), (float*)d_out);
}
